// GPT2Model_55568286876210
// MI455X (gfx1250) — matrix operand order repaired and checked by reading, not compiled, not run on hardware
//
#include <hip/hip_runtime.h>
#include <hip/hip_bf16.h>

// ---------------------------------------------------------------------------
// GPT-2 small forward for gfx1250 (MI455X)
//   - all matmuls: bf16 v_wmma_f32_16x16x32_bf16, fp32 accumulate
//   - global->LDS staging via global_load_async_to_lds_b128 (ASYNCcnt),
//     double-buffered, s_wait_asynccnt pipeline
//   - LN / softmax / residual kept in fp32
// ---------------------------------------------------------------------------

typedef __bf16 bf16_t;
typedef __attribute__((ext_vector_type(16))) __bf16 v16bf;
typedef __attribute__((ext_vector_type(8)))  float  v8f;

#define TB   256      // threads per block (8 waves of 32)
#define BM   128      // block tile M
#define BK   64       // K per LDS tile (2 WMMA K-steps)
#define APAD 16       // -> A row stride 80 el = 160B (16B aligned)
#define BPAD 16       // -> B row stride (BN+16) el, 16B aligned

#define GF_BF16OUT 1
#define GF_RELU    2
#define GF_CAUSAL  4

// ---- CDNA5 async global->LDS copy (16B per lane), ASYNCcnt tracked --------
__device__ __forceinline__ void async_copy_b128(const void* gptr, void* lds_ptr)
{
    // flat pointer into LDS: addr[31:0] is the LDS byte offset (ISA 10.2)
    unsigned lds_off = (unsigned)(unsigned long long)(uintptr_t)lds_ptr;
    unsigned long long gaddr = (unsigned long long)(uintptr_t)gptr;
    asm volatile("global_load_async_to_lds_b128 %0, %1, off"
                 :: "v"(lds_off), "v"(gaddr)
                 : "memory");
}

template <int N>
__device__ __forceinline__ void s_wait_async()
{
    asm volatile("s_wait_asynccnt %0" :: "i"(N) : "memory");
}

// ---------------------------------------------------------------------------
// Batched GEMM: C[z] = A[z] * B[z] (+bias)(+residual), epilogue flags.
// A: [M,K] bf16 row-major (lda), B: [K,N] bf16 row-major (ldb).
// Block tile: 128 x (64*TN), K-step 64. 8 waves: 2 along M x 4 along N.
// Wave tile: 64 x (16*TN). Requires M%128==0, N%(64*TN)==0, K%64==0,
// and 16B alignment of all row starts (true at every call site here).
// ---------------------------------------------------------------------------
template <int TN>
__global__ __launch_bounds__(256)
void gemm_wmma_bf16(const bf16_t* __restrict__ A, const bf16_t* __restrict__ Bm,
                    const float* __restrict__ bias, const float* __restrict__ residual,
                    void* __restrict__ Cout,
                    int M, int N, int K, int lda, int ldb, int ldc,
                    long aBS, long bBS, long cBSo, long cBSi, int cInner,
                    int flags, float scale)
{
    constexpr int BN = 64 * TN;
    __shared__ __align__(16) bf16_t As[2][BM][BK + APAD];
    __shared__ __align__(16) bf16_t Bs[2][BK][BN + BPAD];

    const int tid   = threadIdx.x;
    const int lane  = tid & 31;
    const int wave  = tid >> 5;
    const int waveM = wave & 1;   // 2 waves along M  (64 rows each)
    const int waveN = wave >> 1;  // 4 waves along N  (16*TN cols each)

    const int  z    = blockIdx.z;
    const long aOff = (long)z * aBS;
    const long bOff = (long)z * bBS;
    const long cOff = (long)(z / cInner) * cBSo + (long)(z % cInner) * cBSi;

    const int blockM = blockIdx.y * BM;
    const int blockN = blockIdx.x * BN;

    const int half = lane >> 4;   // A-fragment K-half / C M-half
    const int r    = lane & 15;   // A-fragment row    / C column

    v8f acc[4][TN];
    #pragma unroll
    for (int mi = 0; mi < 4; ++mi)
        #pragma unroll
        for (int ni = 0; ni < TN; ++ni)
            acc[mi][ni] = v8f{};

    // per-tile async issue count per wave: A = 4, B = 2*TN
    constexpr int ISSUES = 4 + 2 * TN;

    const bf16_t* Abase = A + aOff + (long)blockM * lda;
    const bf16_t* Bbase = Bm + bOff + blockN;

    auto load_tile = [&](int kt, int buf) {
        const bf16_t* Ag = Abase + kt * BK;
        #pragma unroll
        for (int i = 0; i < 4; ++i) {                 // 128x64 el = 1024 x 16B
            int c   = i * TB + tid;
            int row = c >> 3;
            int col = (c & 7) << 3;
            async_copy_b128(Ag + (long)row * lda + col, &As[buf][row][col]);
        }
        const bf16_t* Bg = Bbase + (long)(kt * BK) * ldb;
        #pragma unroll
        for (int i = 0; i < 2 * TN; ++i) {            // 64xBN el
            int c   = i * TB + tid;
            int row = c / (8 * TN);
            int col = (c % (8 * TN)) << 3;
            async_copy_b128(Bg + (long)row * ldb + col, &Bs[buf][row][col]);
        }
    };

    auto compute = [&](int buf) {
        #pragma unroll
        for (int kk = 0; kk < BK / 32; ++kk) {
            union Frag { v16bf v; unsigned int u[8]; };
            Frag bfr[TN];
            #pragma unroll
            for (int ni = 0; ni < TN; ++ni)
                #pragma unroll
                for (int j = 0; j < 16; ++j)
                    bfr[ni].v[j] =
                        Bs[buf][kk * 32 + (j & 7) + 8 * half + ((j >= 8) ? 16 : 0)][waveN * 16 * TN + ni * 16 + r];

            #pragma unroll
            for (int mi = 0; mi < 4; ++mi) {
                Frag afr;
                const int arow = waveM * 64 + mi * 16 + r;
                #pragma unroll
                for (int j = 0; j < 8; ++j) {
                    const int kp = 2 * j + 8 * half + ((j >= 4) ? 8 : 0);
                    afr.u[j] = *(const unsigned int*)&As[buf][arow][kk * 32 + kp];
                }
                #pragma unroll
                for (int ni = 0; ni < TN; ++ni)
                    acc[mi][ni] = __builtin_amdgcn_wmma_f32_16x16x32_bf16(
                        false, afr.v, false, bfr[ni].v, (short)0, acc[mi][ni],
                        false, false);
            }
        }
    };

    // ---- double-buffered async pipeline ----
    const int KT = K / BK;
    load_tile(0, 0);
    for (int t = 0; t < KT; ++t) {
        const int buf = t & 1;
        if (t + 1 < KT) {
            load_tile(t + 1, buf ^ 1);
            s_wait_async<ISSUES>();   // tile t fully landed (in-order completion)
        } else {
            s_wait_async<0>();
        }
        __syncthreads();
        compute(buf);
        __syncthreads();
    }

    // ---- epilogue: C element (m = j + 8*half, n = lane&15) ----
    #pragma unroll
    for (int mi = 0; mi < 4; ++mi) {
        #pragma unroll
        for (int ni = 0; ni < TN; ++ni) {
            #pragma unroll
            for (int j = 0; j < 8; ++j) {
                const int mrow = blockM + waveM * 64 + mi * 16 + 8 * half + j;
                const int ncol = blockN + waveN * 16 * TN + ni * 16 + r;
                float v = acc[mi][ni][j];
                if (bias)              v += bias[ncol];
                if (flags & GF_CAUSAL) v = (ncol <= mrow) ? v * scale : -10000.0f;
                if (flags & GF_RELU)   v = v > 0.0f ? v : 0.0f;
                const long off = cOff + (long)mrow * ldc + ncol;
                if (residual)          v += residual[off];
                if (flags & GF_BF16OUT) ((bf16_t*)Cout)[off] = (bf16_t)v;
                else                    ((float*)Cout)[off]  = v;
            }
        }
    }
}

// ---------------------------------------------------------------------------
__global__ void f32_to_bf16(const float* __restrict__ src, bf16_t* __restrict__ dst, long n)
{
    long i      = (long)blockIdx.x * blockDim.x + threadIdx.x;
    long stride = (long)gridDim.x * blockDim.x;
    for (; i < n; i += stride) dst[i] = (bf16_t)src[i];
}

// h[b,s,:] = wte[id[b,s],:] + wpe[s,:]
__global__ void embed_k(const int* __restrict__ ids, const float* __restrict__ wte,
                        const float* __restrict__ wpe, float* __restrict__ h)
{
    long i = (long)blockIdx.x * blockDim.x + threadIdx.x;
    if (i >= (long)4096 * 768) return;
    int row = (int)(i / 768), d = (int)(i % 768);
    int s   = row & 1023;
    int id  = ids[row];
    h[i] = wte[(long)id * 768 + d] + wpe[(long)s * 768 + d];
}

// block-per-row layernorm; writes bf16 (GEMM input) or f32 (final output)
__global__ __launch_bounds__(256)
void layernorm_k(const float* __restrict__ x, const float* __restrict__ g,
                 const float* __restrict__ b, bf16_t* __restrict__ out_bf,
                 float* __restrict__ out_f, int D)
{
    __shared__ float red[256];
    const int row = blockIdx.x;
    const float* xr = x + (long)row * D;

    float s = 0.f, s2 = 0.f;
    for (int d = threadIdx.x; d < D; d += 256) { float v = xr[d]; s += v; s2 += v * v; }

    red[threadIdx.x] = s; __syncthreads();
    for (int o = 128; o > 0; o >>= 1) {
        if (threadIdx.x < o) red[threadIdx.x] += red[threadIdx.x + o];
        __syncthreads();
    }
    const float mean = red[0] / D; __syncthreads();

    red[threadIdx.x] = s2; __syncthreads();
    for (int o = 128; o > 0; o >>= 1) {
        if (threadIdx.x < o) red[threadIdx.x] += red[threadIdx.x + o];
        __syncthreads();
    }
    const float var  = red[0] / D - mean * mean;
    const float rstd = rsqrtf(var + 1e-5f);

    for (int d = threadIdx.x; d < D; d += 256) {
        const float y = (xr[d] - mean) * rstd * g[d] + b[d];
        if (out_bf) out_bf[(long)row * D + d] = (bf16_t)y;
        else        out_f [(long)row * D + d] = y;
    }
}

// qkv f32 [4096,2304] -> q[bh,s,hd] bf16, kT[bh,hd,s] bf16, v[bh,s,hd] bf16
__global__ void split_qkv_k(const float* __restrict__ qkv, bf16_t* __restrict__ q,
                            bf16_t* __restrict__ kT, bf16_t* __restrict__ v)
{
    long i = (long)blockIdx.x * blockDim.x + threadIdx.x;
    if (i >= (long)4096 * 768) return;
    const int row = (int)(i / 768);          // b*S + s
    const int c   = (int)(i % 768);
    const int b   = row >> 10, s = row & 1023;
    const int h   = c >> 6,    hd = c & 63;
    const long base = (long)row * 2304;
    const long zh   = (long)(b * 12 + h);
    q [(zh * 1024 + s) * 64 + hd] = (bf16_t)qkv[base + c];
    kT[(zh * 64 + hd) * 1024 + s] = (bf16_t)qkv[base + 768 + c];
    v [(zh * 1024 + s) * 64 + hd] = (bf16_t)qkv[base + 1536 + c];
}

// row softmax over 1024 (mask already baked in as -1e4); writes bf16 in place
__global__ __launch_bounds__(256)
void softmax_k(float* __restrict__ scores, bf16_t* __restrict__ probs)
{
    __shared__ float red[256];
    const long row = blockIdx.x;
    float*  sr = scores + row * 1024;
    bf16_t* pr = probs  + row * 1024;

    float v[4];
    #pragma unroll
    for (int i = 0; i < 4; ++i) v[i] = sr[threadIdx.x + i * 256];  // all reads first

    float m = fmaxf(fmaxf(v[0], v[1]), fmaxf(v[2], v[3]));
    red[threadIdx.x] = m; __syncthreads();
    for (int o = 128; o > 0; o >>= 1) {
        if (threadIdx.x < o) red[threadIdx.x] = fmaxf(red[threadIdx.x], red[threadIdx.x + o]);
        __syncthreads();
    }
    m = red[0]; __syncthreads();

    float s = 0.f;
    #pragma unroll
    for (int i = 0; i < 4; ++i) { v[i] = __expf(v[i] - m); s += v[i]; }
    red[threadIdx.x] = s; __syncthreads();
    for (int o = 128; o > 0; o >>= 1) {
        if (threadIdx.x < o) red[threadIdx.x] += red[threadIdx.x + o];
        __syncthreads();
    }
    const float inv = 1.f / red[0];

    #pragma unroll
    for (int i = 0; i < 4; ++i)
        pr[threadIdx.x + i * 256] = (bf16_t)(v[i] * inv);          // writes after sync
}

// ---------------------------------------------------------------------------
extern "C" void kernel_launch(void* const* d_in, const int* in_sizes, int n_in,
                              void* d_out, int out_size, void* d_ws, size_t ws_size,
                              hipStream_t stream)
{
    (void)in_sizes; (void)n_in; (void)out_size; (void)ws_size;

    const int*   ids     = (const int*)  d_in[0];
    const float* wte     = (const float*)d_in[1];
    const float* wpe     = (const float*)d_in[2];
    const float* ln1_g   = (const float*)d_in[3];
    const float* ln1_b   = (const float*)d_in[4];
    const float* W_attn  = (const float*)d_in[5];
    const float* b_attn  = (const float*)d_in[6];
    const float* W_aproj = (const float*)d_in[7];
    const float* b_aproj = (const float*)d_in[8];
    const float* ln2_g   = (const float*)d_in[9];
    const float* ln2_b   = (const float*)d_in[10];
    const float* W_fc    = (const float*)d_in[11];
    const float* b_fc    = (const float*)d_in[12];
    const float* W_mproj = (const float*)d_in[13];
    const float* b_mproj = (const float*)d_in[14];
    const float* lnf_g   = (const float*)d_in[15];
    const float* lnf_b   = (const float*)d_in[16];

    const int S = 1024, D = 768, H = 12, F = 3072, L = 12, HD = 64;
    const int M  = 4 * S;       // 4096 rows
    const int BH = 4 * H;       // 48 attention batches

    // ---- workspace bump allocator (all 256B aligned) ----
    char* p = (char*)d_ws;
    auto alloc = [&](size_t bytes) -> void* {
        void* r = (void*)p;
        p += (bytes + 255) & ~(size_t)255;
        return r;
    };
    bf16_t* wattn_bf  = (bf16_t*)alloc((size_t)L * D * 3 * D * 2);
    bf16_t* waproj_bf = (bf16_t*)alloc((size_t)L * D * D * 2);
    bf16_t* wfc_bf    = (bf16_t*)alloc((size_t)L * D * F * 2);
    bf16_t* wmproj_bf = (bf16_t*)alloc((size_t)L * F * D * 2);
    float*  h         = (float*) alloc((size_t)M * D * 4);
    bf16_t* x_bf      = (bf16_t*)alloc((size_t)M * D * 2);
    float*  qkv_f32   = (float*) alloc((size_t)M * 3 * D * 4);
    bf16_t* q_bf      = (bf16_t*)alloc((size_t)BH * S * HD * 2);
    bf16_t* kT_bf     = (bf16_t*)alloc((size_t)BH * HD * S * 2);
    bf16_t* v_bf      = (bf16_t*)alloc((size_t)BH * S * HD * 2);
    float*  scores    = (float*) alloc((size_t)BH * S * S * 4);  // reused as bf16 probs
    bf16_t* attn_bf   = (bf16_t*)alloc((size_t)M * D * 2);
    bf16_t* mlp_bf    = (bf16_t*)alloc((size_t)M * F * 2);

    // ---- weights fp32 -> bf16 (once per call; deterministic) ----
    f32_to_bf16<<<2048, 256, 0, stream>>>(W_attn,  wattn_bf,  (long)L * D * 3 * D);
    f32_to_bf16<<<2048, 256, 0, stream>>>(W_aproj, waproj_bf, (long)L * D * D);
    f32_to_bf16<<<2048, 256, 0, stream>>>(W_fc,    wfc_bf,    (long)L * D * F);
    f32_to_bf16<<<2048, 256, 0, stream>>>(W_mproj, wmproj_bf, (long)L * F * D);

    // ---- embeddings ----
    embed_k<<<(M * D + 255) / 256, 256, 0, stream>>>(ids, wte, wpe, h);

    for (int l = 0; l < L; ++l) {
        const long lD = (long)l * D;

        // x = LN1(h) -> bf16
        layernorm_k<<<M, 256, 0, stream>>>(h, ln1_g + lD, ln1_b + lD, x_bf, nullptr, D);

        // qkv = x @ W_attn + b_attn   [4096,2304] f32
        {
            dim3 g(3 * D / 128, M / BM, 1);
            gemm_wmma_bf16<2><<<g, TB, 0, stream>>>(
                x_bf, wattn_bf + lD * 3 * D, b_attn + (long)l * 3 * D, nullptr, qkv_f32,
                M, 3 * D, D, D, 3 * D, 3 * D, 0L, 0L, 0L, 0L, 1, 0, 1.0f);
        }

        // split to q / kT / v (bf16)
        split_qkv_k<<<(M * D + 255) / 256, 256, 0, stream>>>(qkv_f32, q_bf, kT_bf, v_bf);

        // scores[z] = causal( (q @ kT) / 8 )   48 x [1024,1024] f32
        {
            dim3 g(S / 128, S / BM, BH);
            gemm_wmma_bf16<2><<<g, TB, 0, stream>>>(
                q_bf, kT_bf, nullptr, nullptr, scores,
                S, S, HD, HD, S, S,
                (long)S * HD, (long)HD * S, (long)S * S, 0L, 1,
                GF_CAUSAL, 0.125f);
        }

        // softmax rows -> bf16 probs in place
        softmax_k<<<BH * S, 256, 0, stream>>>(scores, (bf16_t*)scores);

        // attn[b,s,h,:] = probs @ v   (scatter into [4096,768] bf16)
        {
            dim3 g(HD / 64, S / BM, BH);
            gemm_wmma_bf16<1><<<g, TB, 0, stream>>>(
                (const bf16_t*)scores, v_bf, nullptr, nullptr, attn_bf,
                S, HD, S, S, HD, D,
                (long)S * S, (long)S * HD, (long)S * D, (long)HD, H,
                GF_BF16OUT, 1.0f);
        }

        // h = h + attn @ W_aproj + b_aproj
        {
            dim3 g(D / 128, M / BM, 1);
            gemm_wmma_bf16<2><<<g, TB, 0, stream>>>(
                attn_bf, waproj_bf + lD * D, b_aproj + lD, h, h,
                M, D, D, D, D, D, 0L, 0L, 0L, 0L, 1, 0, 1.0f);
        }

        // x = LN2(h) -> bf16
        layernorm_k<<<M, 256, 0, stream>>>(h, ln2_g + lD, ln2_b + lD, x_bf, nullptr, D);

        // m = relu(x @ W_fc + b_fc) -> bf16 [4096,3072]
        {
            dim3 g(F / 128, M / BM, 1);
            gemm_wmma_bf16<2><<<g, TB, 0, stream>>>(
                x_bf, wfc_bf + lD * F, b_fc + (long)l * F, nullptr, mlp_bf,
                M, F, D, D, F, F, 0L, 0L, 0L, 0L, 1,
                GF_BF16OUT | GF_RELU, 1.0f);
        }

        // h = h + m @ W_mproj + b_mproj
        {
            dim3 g(D / 128, M / BM, 1);
            gemm_wmma_bf16<2><<<g, TB, 0, stream>>>(
                mlp_bf, wmproj_bf + (long)l * F * D, b_mproj + lD, h, h,
                M, D, F, F, D, D, 0L, 0L, 0L, 0L, 1, 0, 1.0f);
        }
    }

    // final layernorm -> f32 output
    layernorm_k<<<M, 256, 0, stream>>>(h, lnf_g, lnf_b, nullptr, (float*)d_out, D);
}
